// SlidingWindowTransformer_14396730376912
// MI455X (gfx1250) — compile-verified, matrix operand order checked by reading, not run on hardware
//
#include <hip/hip_runtime.h>

#define Bn    4
#define Sn    512
#define Dn    256
#define Wn    32
#define Ln    4
#define Hn    8
#define DFFn  1024
#define NOUTn 32

typedef __attribute__((ext_vector_type(16))) _Float16 v16h;
typedef __attribute__((ext_vector_type(8)))  _Float16 v8h;
typedef __attribute__((ext_vector_type(8)))  float    v8f;

union HFrag { v16h v; v8h h[2]; };

// Load a 16x32 (MxK) A-fragment or 32x16 (KxN) B-fragment for
// v_wmma_f32_16x16x32_f16 from a row-major [row][k] array (row = M for A,
// row = N for B since weights are stored [n][k] i.e. transposed-B).
__device__ __forceinline__ v16h load_frag(const _Float16* p, int ld, int row0, int k0) {
  const int lane = threadIdx.x & 31;
  const _Float16* q = p + (row0 + (lane & 15)) * ld + k0 + ((lane >> 4) << 3);
  HFrag f;
  f.h[0] = *(const v8h*)(q);
  f.h[1] = *(const v8h*)(q + 16);
  return f.v;
}

__device__ __forceinline__ v8f wmma_f16(v16h a, v16h b, v8f c) {
  return __builtin_amdgcn_wmma_f32_16x16x32_f16(false, a, false, b, (short)0, c, false, false);
}

// D layout: VGPR r, lanes 0-15: (m = row0 + r,     n = col0 + lane)
//                   lanes 16-31:(m = row0 + 8 + r, n = col0 + lane-16)
__device__ __forceinline__ void store_tile_f16(_Float16* p, int ld, int row0, int col0, v8f acc) {
  const int lane = threadIdx.x & 31;
  const int m0 = row0 + ((lane >> 4) << 3);
  _Float16* q = p + col0 + (lane & 15);
#pragma unroll
  for (int r = 0; r < 8; ++r) q[(m0 + r) * ld] = (_Float16)acc[r];
}

// Transposed store: p[n][m]; each lane's 8 elements are m-contiguous -> one b128.
__device__ __forceinline__ void store_tile_f16_T(_Float16* p, int ld, int row0, int col0, v8f acc) {
  const int lane = threadIdx.x & 31;
  const int m0 = row0 + ((lane >> 4) << 3);
  const int n  = col0 + (lane & 15);
  v8h h;
#pragma unroll
  for (int r = 0; r < 8; ++r) h[r] = (_Float16)acc[r];
  *(v8h*)(p + n * ld + m0) = h;
}

// 64 rows, 16 waves -> 4 rows per wave
__device__ __forceinline__ void layer_norm_rows(float* xf32, _Float16* xf16,
                                                const float* __restrict__ g,
                                                const float* __restrict__ bt,
                                                int wave, int lane) {
#pragma unroll
  for (int rr = 0; rr < 4; ++rr) {
    const int m = wave * 4 + rr;
    float vals[8];
    float sum = 0.f;
#pragma unroll
    for (int i = 0; i < 8; ++i) { vals[i] = xf32[m * 256 + lane + 32 * i]; sum += vals[i]; }
#pragma unroll
    for (int o = 16; o >= 1; o >>= 1) sum += __shfl_xor(sum, o, 32);
    const float mu = sum * (1.f / 256.f);
    float s2 = 0.f;
#pragma unroll
    for (int i = 0; i < 8; ++i) { float d = vals[i] - mu; s2 += d * d; }
#pragma unroll
    for (int o = 16; o >= 1; o >>= 1) s2 += __shfl_xor(s2, o, 32);
    const float rstd = rsqrtf(s2 * (1.f / 256.f) + 1e-5f);
#pragma unroll
    for (int i = 0; i < 8; ++i) {
      const int c = lane + 32 * i;
      const float y = (vals[i] - mu) * rstd * g[c] + bt[c];
      xf32[m * 256 + c] = y;
      xf16[m * 272 + c] = (_Float16)y;
    }
  }
}

__global__ void cvt_f32_f16(const float* __restrict__ s, _Float16* __restrict__ d, int n) {
  int i = (blockIdx.x * blockDim.x + threadIdx.x) * 4;
  if (i + 3 < n) {
    float4 v = *(const float4*)(s + i);
    d[i + 0] = (_Float16)v.x;
    d[i + 1] = (_Float16)v.y;
    d[i + 2] = (_Float16)v.z;
    d[i + 3] = (_Float16)v.w;
  }
}

// 2 windows per workgroup, 16 waves. M-tiles: 0,1 = window0 rows; 2,3 = window1 rows.
// Every wave processes all 4 M-tiles per B-fragment (4x weight reuse) and multiple
// N-tiles per A-fragment (2-3x LDS-read reuse).
__global__ __launch_bounds__(512) void swt_kernel(
    const float* __restrict__ X, const float* __restrict__ pos,
    const _Float16* __restrict__ Win,  const float* __restrict__ Bin,
    const _Float16* __restrict__ Wout, const float* __restrict__ Bout,
    const _Float16* __restrict__ W1,   const float* __restrict__ B1,
    const _Float16* __restrict__ W2,   const float* __restrict__ B2,
    const float* __restrict__ ln1g, const float* __restrict__ ln1b,
    const float* __restrict__ ln2g, const float* __restrict__ ln2b,
    const float* __restrict__ headw, const float* __restrict__ headb,
    float* __restrict__ out) {
  // LDS ~244KB -> 1 WG per WGP (320KB), 16 waves = 4 per SIMD
  __shared__ float    xf32[64 * 256];     // master activations (2 windows)
  __shared__ _Float16 l16[92160];
  _Float16* xf16 = l16;                   // [64][272]
  _Float16* qk   = l16 + 17408;           // [64][528]: q 0..255, k 256..511 (reused: attn-out / ffn hidden 512)
  _Float16* vT   = l16 + 51200;           // [2][256][40]: v transposed per window
  _Float16* pr   = l16 + 71680;           // [16][32][40]: per-wave probability scratch

  const int tid  = threadIdx.x;
  const int lane = tid & 31;
  const int wave = __builtin_amdgcn_readfirstlane(tid >> 5);  // 0..15, scalar
  const int pair0 = blockIdx.x * 2;       // first window index
  const int bb   = pair0 >> 9;
  const int ss0  = pair0 & 511;

  // ---- gather 2 sliding windows + pos embed (64 rows, 8 threads/row) ----
  {
    const int jr = tid >> 3;              // 0..63
    const int wwin = jr >> 5, j = jr & 31;
    const int c0 = (tid & 7) * 32;
    int src = (ss0 + wwin) - j; if (src < 0) src = 0;
    const float* xr = X + ((bb << 9) + src) * 256;
    const float* pe = pos + j * 256;
#pragma unroll
    for (int u = 0; u < 8; ++u) {
      const int c = c0 + u * 4;
      float4 xv = *(const float4*)(xr + c);
      float4 pv = *(const float4*)(pe + c);
      float y0 = xv.x + pv.x, y1 = xv.y + pv.y, y2 = xv.z + pv.z, y3 = xv.w + pv.w;
      xf32[jr * 256 + c + 0] = y0; xf16[jr * 272 + c + 0] = (_Float16)y0;
      xf32[jr * 256 + c + 1] = y1; xf16[jr * 272 + c + 1] = (_Float16)y1;
      xf32[jr * 256 + c + 2] = y2; xf16[jr * 272 + c + 2] = (_Float16)y2;
      xf32[jr * 256 + c + 3] = y3; xf16[jr * 272 + c + 3] = (_Float16)y3;
    }
  }
  __syncthreads();

  const v8f vzero = {};
  const int m0l = ((lane >> 4) << 3);
  const int nnl = lane & 15;
  const int kh  = wave >> 3;              // split-K half for out-proj / ffn2
  const int np  = wave & 7;               // N-tile-pair id for out-proj / ffn2

  for (int l = 0; l < Ln; ++l) {
    const _Float16* wqkv = Win  + l * 768 * 256;
    const float*    bqkv = Bin  + l * 768;
    const _Float16* wo   = Wout + l * 256 * 256;
    const float*    bo   = Bout + l * 256;
    const _Float16* w1l  = W1   + l * 1024 * 256;
    const float*    b1l  = B1   + l * 1024;
    const _Float16* w2l  = W2   + l * 256 * 1024;
    const float*    b2l  = B2   + l * 256;

    // ---- QKV GEMM: wave owns column tile n0 of Q, K and V (Nw=3, 12 wmma / 4 A-frags) ----
    {
      const int n0 = wave * 16;           // scalar
      v8f acc[4][3];
#pragma unroll
      for (int mi = 0; mi < 4; ++mi) { acc[mi][0] = vzero; acc[mi][1] = vzero; acc[mi][2] = vzero; }
#pragma unroll
      for (int kt = 0; kt < 8; ++kt) {
        v16h bq = load_frag(wqkv, 256, n0,       kt * 32);
        v16h bk = load_frag(wqkv, 256, n0 + 256, kt * 32);
        v16h bv = load_frag(wqkv, 256, n0 + 512, kt * 32);
#pragma unroll
        for (int mi = 0; mi < 4; ++mi) {
          v16h a = load_frag(xf16, 272, mi * 16, kt * 32);
          acc[mi][0] = wmma_f16(a, bq, acc[mi][0]);
          acc[mi][1] = wmma_f16(a, bk, acc[mi][1]);
          acc[mi][2] = wmma_f16(a, bv, acc[mi][2]);
        }
      }
      const float biasq = bqkv[n0 + nnl];
      const float biask = bqkv[n0 + 256 + nnl];
      const float biasv = bqkv[n0 + 512 + nnl];
#pragma unroll
      for (int mi = 0; mi < 4; ++mi) {
#pragma unroll
        for (int r = 0; r < 8; ++r) {
          acc[mi][0][r] += biasq; acc[mi][1][r] += biask; acc[mi][2][r] += biasv;
        }
        store_tile_f16(qk, 528, mi * 16, n0,       acc[mi][0]);                        // Q
        store_tile_f16(qk, 528, mi * 16, n0 + 256, acc[mi][1]);                        // K
        store_tile_f16_T(vT + (mi >> 1) * 10240, 40, (mi & 1) * 16, n0, acc[mi][2]);   // V^T
      }
    }
    __syncthreads();

    // ---- attention: 16 (window, head) pairs on 16 waves ----
    {
      const int aw = wave >> 3;             // window
      const int h  = wave & 7;              // head
      const int rb = aw * 32;               // row base
      _Float16* vTw = vT + aw * 10240;
      _Float16* ph  = pr + wave * 1280;
      v8f sc[2][2];
#pragma unroll
      for (int qt = 0; qt < 2; ++qt) {
        v16h a = load_frag(qk, 528, rb + qt * 16, h * 32);                // Q rows
#pragma unroll
        for (int nt = 0; nt < 2; ++nt) {
          v16h b = load_frag(qk + 256, 528, rb + nt * 16, h * 32);       // K rows as B
          sc[qt][nt] = wmma_f16(a, b, vzero);
        }
      }
      const float scale = 0.17677669529663687f; // 32^-0.5
#pragma unroll
      for (int qt = 0; qt < 2; ++qt) {
#pragma unroll
        for (int r = 0; r < 8; ++r) {
          float s0 = sc[qt][0][r] * scale;
          float s1 = sc[qt][1][r] * scale;
          float mx = fmaxf(s0, s1);
#pragma unroll
          for (int o = 8; o >= 1; o >>= 1) mx = fmaxf(mx, __shfl_xor(mx, o, 32));
          const float e0 = __expf(s0 - mx), e1 = __expf(s1 - mx);
          float sm = e0 + e1;
#pragma unroll
          for (int o = 8; o >= 1; o >>= 1) sm += __shfl_xor(sm, o, 32);
          const float inv = 1.0f / sm;
          const int m = qt * 16 + m0l + r;
          ph[m * 40 + nnl]      = (_Float16)(e0 * inv);
          ph[m * 40 + 16 + nnl] = (_Float16)(e1 * inv);
        }
      }
      // o = probs @ V : write into own Q slice (rows rb.., cols h*32..)
#pragma unroll
      for (int qt = 0; qt < 2; ++qt) {
        v16h a = load_frag(ph, 40, qt * 16, 0);
#pragma unroll
        for (int dt = 0; dt < 2; ++dt) {
          v16h b = load_frag(vTw, 40, h * 32 + dt * 16, 0);
          v8f oa = wmma_f16(a, b, vzero);
          store_tile_f16(qk, 528, rb + qt * 16, h * 32 + dt * 16, oa);
        }
      }
    }
    __syncthreads();

    // ---- out projection: split-K x2, Nw=2; combine halves via LDS f32 atomics ----
    {
      const int n0 = np * 32;               // two N-tiles: n0, n0+16
      v8f acc[4][2];
#pragma unroll
      for (int mi = 0; mi < 4; ++mi) { acc[mi][0] = vzero; acc[mi][1] = vzero; }
#pragma unroll
      for (int kk = 0; kk < 4; ++kk) {
        const int kt = kh * 4 + kk;
        v16h b0 = load_frag(wo, 256, n0,      kt * 32);
        v16h b1 = load_frag(wo, 256, n0 + 16, kt * 32);
#pragma unroll
        for (int mi = 0; mi < 4; ++mi) {
          v16h a = load_frag(qk, 528, mi * 16, kt * 32);
          acc[mi][0] = wmma_f16(a, b0, acc[mi][0]);
          acc[mi][1] = wmma_f16(a, b1, acc[mi][1]);
        }
      }
      const float bias0 = (kh == 0) ? bo[n0 + nnl] : 0.f;
      const float bias1 = (kh == 0) ? bo[n0 + 16 + nnl] : 0.f;
#pragma unroll
      for (int mi = 0; mi < 4; ++mi) {
        const int m0 = mi * 16 + m0l;
#pragma unroll
        for (int r = 0; r < 8; ++r) {
          atomicAdd(&xf32[(m0 + r) * 256 + n0 + nnl],      acc[mi][0][r] + bias0);
          atomicAdd(&xf32[(m0 + r) * 256 + n0 + 16 + nnl], acc[mi][1][r] + bias1);
        }
      }
    }
    __syncthreads();
    layer_norm_rows(xf32, xf16, ln1g + l * 256, ln1b + l * 256, wave, lane);
    __syncthreads();

    // ---- FFN: hidden (1024) in 2 super-chunks of 512 via LDS ----
    v8f facc[4][2];
#pragma unroll
    for (int mi = 0; mi < 4; ++mi) { facc[mi][0] = vzero; facc[mi][1] = vzero; }
    for (int cp = 0; cp < 2; ++cp) {
      {
        // hidden: wave owns one tile in each half of the super-chunk (Nw=2)
        const int ng0 = cp * 512 + wave * 16;         // global hidden rows of w1
        const int ng1 = ng0 + 256;
        v8f acc[4][2];
#pragma unroll
        for (int mi = 0; mi < 4; ++mi) { acc[mi][0] = vzero; acc[mi][1] = vzero; }
#pragma unroll
        for (int kt = 0; kt < 8; ++kt) {
          v16h b0 = load_frag(w1l, 256, ng0, kt * 32);
          v16h b1 = load_frag(w1l, 256, ng1, kt * 32);
#pragma unroll
          for (int mi = 0; mi < 4; ++mi) {
            v16h a = load_frag(xf16, 272, mi * 16, kt * 32);
            acc[mi][0] = wmma_f16(a, b0, acc[mi][0]);
            acc[mi][1] = wmma_f16(a, b1, acc[mi][1]);
          }
        }
        const float bias0 = b1l[ng0 + nnl];
        const float bias1 = b1l[ng1 + nnl];
#pragma unroll
        for (int mi = 0; mi < 4; ++mi) {
#pragma unroll
          for (int r = 0; r < 8; ++r) {
            acc[mi][0][r] = fmaxf(acc[mi][0][r] + bias0, 0.f);   // ReLU
            acc[mi][1][r] = fmaxf(acc[mi][1][r] + bias1, 0.f);
          }
          store_tile_f16(qk, 528, mi * 16, wave * 16,       acc[mi][0]);
          store_tile_f16(qk, 528, mi * 16, 256 + wave * 16, acc[mi][1]);
        }
      }
      __syncthreads();
      {
        // consume 512-wide hidden: split-K x2 (8 of 16 k-tiles per wave), Nw=2
        const int n0 = np * 32;
#pragma unroll
        for (int kk = 0; kk < 8; ++kk) {
          const int kt = kh * 8 + kk;
          v16h b0 = load_frag(w2l, 1024, n0,      cp * 512 + kt * 32);
          v16h b1 = load_frag(w2l, 1024, n0 + 16, cp * 512 + kt * 32);
#pragma unroll
          for (int mi = 0; mi < 4; ++mi) {
            v16h a = load_frag(qk, 528, mi * 16, kt * 32);
            facc[mi][0] = wmma_f16(a, b0, facc[mi][0]);
            facc[mi][1] = wmma_f16(a, b1, facc[mi][1]);
          }
        }
      }
      __syncthreads();
    }
    {
      const int n0 = np * 32;
      const float bias0 = (kh == 0) ? b2l[n0 + nnl] : 0.f;
      const float bias1 = (kh == 0) ? b2l[n0 + 16 + nnl] : 0.f;
#pragma unroll
      for (int mi = 0; mi < 4; ++mi) {
        const int m0 = mi * 16 + m0l;
#pragma unroll
        for (int r = 0; r < 8; ++r) {
          atomicAdd(&xf32[(m0 + r) * 256 + n0 + nnl],      facc[mi][0][r] + bias0);
          atomicAdd(&xf32[(m0 + r) * 256 + n0 + 16 + nnl], facc[mi][1][r] + bias1);
        }
      }
    }
    __syncthreads();
    layer_norm_rows(xf32, xf16, ln2g + l * 256, ln2b + l * 256, wave, lane);
    __syncthreads();
  }

  // ---- head: last window slot (row 31 of each window), waves 0 and 1 ----
  if (wave < 2) {
    float acc = headb[lane];
    const float* hw = headw + lane * 256;
    const float* xr = xf32 + (wave * 32 + 31) * 256;
    for (int k = 0; k < 256; k += 4) {
      float4 w4 = *(const float4*)(hw + k);
      acc += xr[k] * w4.x + xr[k + 1] * w4.y + xr[k + 2] * w4.z + xr[k + 3] * w4.w;
    }
    out[(pair0 + wave) * 32 + lane] = acc;
  }
}

extern "C" void kernel_launch(void* const* d_in, const int* in_sizes, int n_in,
                              void* d_out, int out_size, void* d_ws, size_t ws_size,
                              hipStream_t stream) {
  const float* X     = (const float*)d_in[0];
  const float* pos   = (const float*)d_in[3];
  const float* w_in  = (const float*)d_in[4];
  const float* b_in  = (const float*)d_in[5];
  const float* w_out = (const float*)d_in[6];
  const float* b_out = (const float*)d_in[7];
  const float* w1    = (const float*)d_in[8];
  const float* b1    = (const float*)d_in[9];
  const float* w2    = (const float*)d_in[10];
  const float* b2    = (const float*)d_in[11];
  const float* ln1g  = (const float*)d_in[12];
  const float* ln1b  = (const float*)d_in[13];
  const float* ln2g  = (const float*)d_in[14];
  const float* ln2b  = (const float*)d_in[15];
  const float* headw = (const float*)d_in[16];
  const float* headb = (const float*)d_in[17];
  float* out = (float*)d_out;

  // f16 weight images in workspace (6.3 MB, L2-resident)
  _Float16* ws      = (_Float16*)d_ws;
  _Float16* w_in_h  = ws;
  _Float16* w_out_h = w_in_h  + Ln * 768 * 256;
  _Float16* w1_h    = w_out_h + Ln * 256 * 256;
  _Float16* w2_h    = w1_h    + Ln * 1024 * 256;

  const int n_in_w  = Ln * 768 * 256;
  const int n_out_w = Ln * 256 * 256;
  const int n_w1    = Ln * 1024 * 256;
  const int n_w2    = Ln * 256 * 1024;
  cvt_f32_f16<<<n_in_w  / 1024, 256, 0, stream>>>(w_in,  w_in_h,  n_in_w);
  cvt_f32_f16<<<n_out_w / 1024, 256, 0, stream>>>(w_out, w_out_h, n_out_w);
  cvt_f32_f16<<<n_w1    / 1024, 256, 0, stream>>>(w1,    w1_h,    n_w1);
  cvt_f32_f16<<<n_w2    / 1024, 256, 0, stream>>>(w2,    w2_h,    n_w2);

  swt_kernel<<<(Bn * Sn) / 2, 512, 0, stream>>>(X, pos, w_in_h, b_in, w_out_h, b_out,
                                                w1_h, b1, w2_h, b2, ln1g, ln1b, ln2g,
                                                ln2b, headw, headb, out);
}
